// ISA_34892314313623
// MI455X (gfx1250) — matrix operand order repaired and checked by reading, compiled, not run on hardware
//
#include <hip/hip_runtime.h>
#include <math.h>

// ---------------------------------------------------------------------------
// Transposed channel attention, algebraically collapsed:
//   G[b]   = X[b] X[b]^T                   (X[b] = x[b] viewed as [256,16384])
//   scores = f(Wq G Wk^T, norms from diag terms, Sx row sums, biases)
//   attn   = softmax(scores)
//   y[b]   = (proj_w @ attn @ Wv) @ X[b] + (proj_w @ attn @ bv + proj_b)
// Two streaming passes over x (WMMA f16 x f16 -> f32), tiny f32 GEMMs between.
// WMMA passes use a 2-deep software pipeline: loads for chunk it+2 are issued
// in iteration it, so their wait lands a full iteration after issue.
// ---------------------------------------------------------------------------

#define CH 256
#define HW 16384
#define NB 8
#define EPSN 1e-12f
#define KSLICES 8
#define KSL (HW / KSLICES)   // 2048 per slice

typedef _Float16 v16h __attribute__((ext_vector_type(16)));
typedef float    v8f  __attribute__((ext_vector_type(8)));

// Build 16x32 f16 A-fragment. Lane L: row = L&15, h = L>>4.
// elements 0..7  = K[8h .. 8h+8)   (VGPRs 0-3)
// elements 8..15 = K[16+8h .. +8)  (VGPRs 4-7)
__device__ __forceinline__ v16h load_a_frag(const _Float16* rowptr, int h) {
    v16h a;
    const _Float16* p0 = rowptr + 8 * h;
    const _Float16* p1 = rowptr + 16 + 8 * h;
#pragma unroll
    for (int j = 0; j < 8; ++j) { a[j] = p0[j]; a[8 + j] = p1[j]; }
    return a;
}

// Build 32x16 f16 B-fragment. Lane L: col = L&15, h = L>>4.
// elements 0..7  = K[8h .. 8h+8), elements 8..15 = K[16+8h .. +8)
__device__ __forceinline__ v16h load_b_frag(const _Float16* colptr, int h) {
    v16h b;
    const _Float16* p0 = colptr + 8 * h;
    const _Float16* p1 = colptr + 16 + 8 * h;
#pragma unroll
    for (int j = 0; j < 8; ++j) { b[j] = p0[j]; b[8 + j] = p1[j]; }
    return b;
}

__device__ __forceinline__ void store4h(_Float16* p, float4 v) {
    p[0] = (_Float16)v.x; p[1] = (_Float16)v.y;
    p[2] = (_Float16)v.z; p[3] = (_Float16)v.w;
}

// ---------------------------------------------------------------------------
// Kernel 1: Sx[b*256+c] = sum_s x[b,c,s]
// ---------------------------------------------------------------------------
__global__ void rowsum_kernel(const float* __restrict__ x, float* __restrict__ Sx) {
    __shared__ float red[256];
    int rc = blockIdx.x;                       // b*256 + c
    const float* p = x + (size_t)rc * HW;
    float s = 0.f;
    for (int i = threadIdx.x; i < HW; i += 256) s += p[i];
    red[threadIdx.x] = s;
    __syncthreads();
    for (int st = 128; st > 0; st >>= 1) {
        if (threadIdx.x < st) red[threadIdx.x] += red[threadIdx.x + st];
        __syncthreads();
    }
    if (threadIdx.x == 0) Sx[rc] = red[0];
}

// ---------------------------------------------------------------------------
// Kernel 2: partial Gram  Gpart[b*8+sl] = X[b][:, sl*2048 : +2048] @ (same)^T
// Grid (4,4,64) = (tn, tm, b*8+sl), block 128 (4 waves): 64x64 tile.
// 2-deep pipeline: iteration it computes chunk it, stores chunk it+1 to the
// other LDS buffer (load-wait lands here, one iteration after issue), and
// issues global loads for chunk it+2.
// ---------------------------------------------------------------------------
__global__ void gram_kernel(const float* __restrict__ x, float* __restrict__ Gpart) {
    __shared__ _Float16 As[2][64][40];   // [buf][row][k], 80B row stride
    __shared__ _Float16 Bs[2][64][40];
    const int z  = blockIdx.z;
    const int b  = z >> 3;
    const int sl = z & 7;
    const int tm = blockIdx.y, tn = blockIdx.x;
    const float* Xb = x + (size_t)b * CH * HW;
    const int kbeg = sl * KSL;

    const int t    = threadIdx.x;       // 0..127
    const int lane = t & 31;
    const int w    = t >> 5;            // wave 0..3
    const int h    = lane >> 4;
    const int l16  = lane & 15;
    const int lr   = t >> 1;            // staged row 0..63
    const int lc   = (t & 1) * 16;      // half-row offset

    const float* pa = Xb + (size_t)(tm * 64 + lr) * HW + lc + kbeg;
    const float* pb = Xb + (size_t)(tn * 64 + lr) * HW + lc + kbeg;

    v8f acc[4] = {};
    float4 ra[4], rb[4];

    // prologue: chunk 0 -> regs -> buf 0; chunk 1 -> regs
#pragma unroll
    for (int q = 0; q < 4; ++q) {
        ra[q] = ((const float4*)pa)[q];
        rb[q] = ((const float4*)pb)[q];
    }
#pragma unroll
    for (int q = 0; q < 4; ++q) {
        store4h(&As[0][lr][lc + q * 4], ra[q]);
        store4h(&Bs[0][lr][lc + q * 4], rb[q]);
    }
#pragma unroll
    for (int q = 0; q < 4; ++q) {
        ra[q] = ((const float4*)(pa + 32))[q];
        rb[q] = ((const float4*)(pb + 32))[q];
    }

    const int NIT = KSL / 32;           // 64
    for (int it = 0; it < NIT; ++it) {
        __syncthreads();                // buf[it&1] visible to all waves
        const int buf = it & 1;

        if (it + 6 < NIT) {             // long-range prefetch into GL2
            __builtin_prefetch(pa + (it + 6) * 32, 0, 1);
            __builtin_prefetch(pb + (it + 6) * 32, 0, 1);
        }

        // compute on current buffer (loads for chunk it+1 already in flight)
        v16h afrag = load_a_frag(&As[buf][w * 16 + l16][0], h);
#pragma unroll
        for (int nt = 0; nt < 4; ++nt) {
            v16h bfrag = load_b_frag(&Bs[buf][nt * 16 + l16][0], h);
            acc[nt] = __builtin_amdgcn_wmma_f32_16x16x32_f16(
                false, afrag, false, bfrag, (short)0, acc[nt], false, false);
        }

        if (it + 1 < NIT) {             // chunk it+1: wait + convert + store
#pragma unroll
            for (int q = 0; q < 4; ++q) {
                store4h(&As[buf ^ 1][lr][lc + q * 4], ra[q]);
                store4h(&Bs[buf ^ 1][lr][lc + q * 4], rb[q]);
            }
        }
        if (it + 2 < NIT) {             // issue loads for chunk it+2
#pragma unroll
            for (int q = 0; q < 4; ++q) {
                ra[q] = ((const float4*)(pa + (it + 2) * 32))[q];
                rb[q] = ((const float4*)(pb + (it + 2) * 32))[q];
            }
        }
    }

    // C/D layout: VGPR i -> M = i + 8h, N = lane&15
    float* Gp = Gpart + (size_t)z * CH * CH;
#pragma unroll
    for (int nt = 0; nt < 4; ++nt)
#pragma unroll
        for (int i = 0; i < 8; ++i) {
            int row = tm * 64 + w * 16 + i + 8 * h;
            int col = tn * 64 + nt * 16 + l16;
            Gp[(size_t)row * CH + col] = acc[nt][i];
        }
}

// G[b] = sum_sl Gpart[b*8+sl]  (deterministic split-K reduction)
__global__ void greduce_kernel(const float* __restrict__ Gpart, float* __restrict__ G) {
    int e = blockIdx.x * 256 + threadIdx.x;     // 0 .. NB*CH*CH-1
    int b = e >> 16, i = e & 65535;
    float s = 0.f;
#pragma unroll
    for (int sl = 0; sl < KSLICES; ++sl)
        s += Gpart[(((size_t)(b * KSLICES + sl)) << 16) + i];
    G[e] = s;
}

// ---------------------------------------------------------------------------
// Small f32 GEMMs (O(C^3), ~1% of total work): C = A @ B  (nn) and A @ B^T (nt)
// ---------------------------------------------------------------------------
__global__ void gemm_nn_kernel(const float* __restrict__ A, const float* __restrict__ B,
                               float* __restrict__ Cc, int K, int lda, int ldb, int ldc,
                               long Ab, long Bb, long Cb) {
    __shared__ float At[16][17];
    __shared__ float Bt[16][17];
    A  += (long)blockIdx.z * Ab;
    B  += (long)blockIdx.z * Bb;
    Cc += (long)blockIdx.z * Cb;
    const int ty = threadIdx.y, tx = threadIdx.x;
    const int row = blockIdx.y * 16 + ty, col = blockIdx.x * 16 + tx;
    float acc = 0.f;
    for (int k0 = 0; k0 < K; k0 += 16) {
        At[ty][tx] = A[(size_t)row * lda + k0 + tx];
        Bt[ty][tx] = B[(size_t)(k0 + ty) * ldb + col];
        __syncthreads();
#pragma unroll
        for (int kk = 0; kk < 16; ++kk) acc += At[ty][kk] * Bt[kk][tx];
        __syncthreads();
    }
    Cc[(size_t)row * ldc + col] = acc;
}

__global__ void gemm_nt_kernel(const float* __restrict__ A, const float* __restrict__ B,
                               float* __restrict__ Cc, int K, int lda, int ldb, int ldc,
                               long Ab, long Bb, long Cb) {
    __shared__ float At[16][17];
    __shared__ float Bt[16][17];
    A  += (long)blockIdx.z * Ab;
    B  += (long)blockIdx.z * Bb;
    Cc += (long)blockIdx.z * Cb;
    const int ty = threadIdx.y, tx = threadIdx.x;
    const int row = blockIdx.y * 16 + ty, col = blockIdx.x * 16 + tx;
    float acc = 0.f;
    for (int k0 = 0; k0 < K; k0 += 16) {
        At[ty][tx] = A[(size_t)row * lda + k0 + tx];
        Bt[ty][tx] = B[(size_t)(blockIdx.x * 16 + ty) * ldb + k0 + tx];
        __syncthreads();
#pragma unroll
        for (int kk = 0; kk < 16; ++kk) acc += At[ty][kk] * Bt[tx][kk];
        __syncthreads();
    }
    Cc[(size_t)row * ldc + col] = acc;
}

// ---------------------------------------------------------------------------
// norms + W.Sx for rows 0..511 (Wq rows then Wk rows).
// |q_i|^2 = Wi G Wi^T + 2 b_i (Wi.Sx) + b_i^2 * HW, using WqkG = [Wq;Wk] @ G.
// ---------------------------------------------------------------------------
__global__ void norms_kernel(const float* __restrict__ WqkG, const float* __restrict__ qkv_w,
                             const float* __restrict__ qkv_b, const float* __restrict__ Sx,
                             float* __restrict__ norms, float* __restrict__ vvec) {
    __shared__ float r1[256];
    __shared__ float r2[256];
    const int i = blockIdx.x, b = blockIdx.y, t = threadIdx.x;
    float wv = qkv_w[(size_t)i * CH + t];
    r1[t] = wv * WqkG[((size_t)b * 512 + i) * CH + t];
    r2[t] = wv * Sx[b * CH + t];
    __syncthreads();
    for (int st = 128; st > 0; st >>= 1) {
        if (t < st) { r1[t] += r1[t + st]; r2[t] += r2[t + st]; }
        __syncthreads();
    }
    if (t == 0) {
        float v  = r2[0];
        float bi = qkv_b[i];
        float n2 = r1[0] + 2.f * bi * v + bi * bi * (float)HW;
        norms[b * 512 + i] = sqrtf(fmaxf(n2, 0.f));
        vvec[b * 512 + i]  = v;
    }
}

// ---------------------------------------------------------------------------
// scores (bias + normalization + 1/sqrt(HW) scale) and row softmax.
// ---------------------------------------------------------------------------
__global__ void softmax_kernel(const float* __restrict__ QKraw, const float* __restrict__ norms,
                               const float* __restrict__ vvec, const float* __restrict__ qkv_b,
                               float* __restrict__ attn) {
    __shared__ float red[256];
    const int o = blockIdx.x, b = blockIdx.y, p = threadIdx.x;
    float qn = fmaxf(norms[b * 512 + o], EPSN);
    float kn = fmaxf(norms[b * 512 + 256 + p], EPSN);
    float vq = vvec[b * 512 + o];
    float vk = vvec[b * 512 + 256 + p];
    float bq = qkv_b[o];
    float bk = qkv_b[256 + p];
    float raw = QKraw[((size_t)b * CH + o) * CH + p] + bq * vk + bk * vq + bq * bk * (float)HW;
    float s = raw / (qn * kn) * (1.0f / 128.0f);   // scale = 1/sqrt(16384)

    red[p] = s; __syncthreads();
    for (int st = 128; st > 0; st >>= 1) {
        if (p < st) red[p] = fmaxf(red[p], red[p + st]);
        __syncthreads();
    }
    float m = red[0]; __syncthreads();
    float e = expf(s - m);
    red[p] = e; __syncthreads();
    for (int st = 128; st > 0; st >>= 1) {
        if (p < st) red[p] += red[p + st];
        __syncthreads();
    }
    attn[((size_t)b * CH + o) * CH + p] = e / red[0];
}

// beff[b][o] = sum_d M1[b][o][d] * bv[d] + proj_b[o]
__global__ void beff_kernel(const float* __restrict__ M1, const float* __restrict__ qkv_b,
                            const float* __restrict__ proj_b, float* __restrict__ beff) {
    const int b = blockIdx.x, o = threadIdx.x;
    const float* row = M1 + ((size_t)b * CH + o) * CH;
    float s = 0.f;
    for (int d = 0; d < CH; ++d) s += row[d] * qkv_b[512 + d];
    beff[b * CH + o] = s + proj_b[o];
}

// ---------------------------------------------------------------------------
// y[b] = Meff[b] @ X[b] + beff[b]   (pass 2 over x; WMMA; 2-deep pipeline)
// Grid (128, 4, 8) = (N/128, M/64, b), block 256 (8 waves in a 4x2 grid).
// Meff 64x256 staged once; X staged 32x128 per step, transposed to [n][k].
// ---------------------------------------------------------------------------
__global__ void out_kernel(const float* __restrict__ Meff, const float* __restrict__ x,
                           const float* __restrict__ beff, float* __restrict__ y) {
    __shared__ _Float16 As[64][264];     // [m][k], pad 8: 528B row stride
    __shared__ _Float16 Bs[2][128][40];  // [buf][n][k]
    const int b  = blockIdx.z;
    const int n0 = blockIdx.x * 128;
    const int m0 = blockIdx.y * 64;
    const float* Mb = Meff + (size_t)b * CH * CH;
    const float* Xb = x + (size_t)b * CH * HW;

    const int t    = threadIdx.x;
    const int lane = t & 31;
    const int w    = t >> 5;
    const int wm   = w & 3;          // wave M index (4)
    const int wn   = w >> 2;         // wave N index (2)
    const int h    = lane >> 4;
    const int l16  = lane & 15;

    // stage Meff tile once: 64 rows x 256 K (4096 float4 / 256 threads)
    for (int i = t; i < 64 * 64; i += 256) {
        int r  = i >> 6;
        int c4 = (i & 63) * 4;
        float4 v = *(const float4*)(Mb + (size_t)(m0 + r) * CH + c4);
        store4h(&As[r][c4], v);
    }

    // per-thread staging coordinates for X chunks (32 k x 128 n, 4 float4 each)
    const int kk0 = t >> 5;          // base k row (+ 8*q)
    const int cc4 = (t & 31) * 4;    // n offset
    const float* xb0 = Xb + (size_t)kk0 * HW + n0 + cc4;

    float4 rb[4];
    // prologue: chunk 0 -> regs -> Bs[0]; chunk 1 -> regs
#pragma unroll
    for (int q = 0; q < 4; ++q)
        rb[q] = *(const float4*)(xb0 + (size_t)(q * 8) * HW);
#pragma unroll
    for (int q = 0; q < 4; ++q) {
        float4 v = rb[q];
        int k = kk0 + q * 8;
        Bs[0][cc4 + 0][k] = (_Float16)v.x;
        Bs[0][cc4 + 1][k] = (_Float16)v.y;
        Bs[0][cc4 + 2][k] = (_Float16)v.z;
        Bs[0][cc4 + 3][k] = (_Float16)v.w;
    }
#pragma unroll
    for (int q = 0; q < 4; ++q)
        rb[q] = *(const float4*)(xb0 + (size_t)(32 + q * 8) * HW);

    v8f acc[4] = {};
    const int NIT = CH / 32;         // 8
    for (int it = 0; it < NIT; ++it) {
        __syncthreads();
        const int buf = it & 1;

        const int k0 = it * 32;
        v16h afrag = load_a_frag(&As[wm * 16 + l16][k0], h);
#pragma unroll
        for (int nt = 0; nt < 4; ++nt) {
            v16h bfrag = load_b_frag(&Bs[buf][wn * 64 + nt * 16 + l16][0], h);
            acc[nt] = __builtin_amdgcn_wmma_f32_16x16x32_f16(
                false, afrag, false, bfrag, (short)0, acc[nt], false, false);
        }

        if (it + 1 < NIT) {          // chunk it+1: wait + convert + store
#pragma unroll
            for (int q = 0; q < 4; ++q) {
                float4 v = rb[q];
                int k = kk0 + q * 8;
                Bs[buf ^ 1][cc4 + 0][k] = (_Float16)v.x;
                Bs[buf ^ 1][cc4 + 1][k] = (_Float16)v.y;
                Bs[buf ^ 1][cc4 + 2][k] = (_Float16)v.z;
                Bs[buf ^ 1][cc4 + 3][k] = (_Float16)v.w;
            }
        }
        if (it + 2 < NIT) {          // issue loads for chunk it+2
#pragma unroll
            for (int q = 0; q < 4; ++q)
                rb[q] = *(const float4*)(xb0 + (size_t)((it + 2) * 32 + q * 8) * HW);
        }
    }

#pragma unroll
    for (int nt = 0; nt < 4; ++nt)
#pragma unroll
        for (int i = 0; i < 8; ++i) {
            int row = m0 + wm * 16 + i + 8 * h;
            int col = n0 + wn * 64 + nt * 16 + l16;
            y[((size_t)b * CH + row) * HW + col] = acc[nt][i] + beff[b * CH + row];
        }
}

// ---------------------------------------------------------------------------
extern "C" void kernel_launch(void* const* d_in, const int* in_sizes, int n_in,
                              void* d_out, int out_size, void* d_ws, size_t ws_size,
                              hipStream_t stream) {
    const float* x      = (const float*)d_in[0];  // [8,256,128,128]
    // d_in[1] = temperature: dead code in reference (attn branch unused)
    const float* qkv_w  = (const float*)d_in[2];  // [768,256]
    const float* qkv_b  = (const float*)d_in[3];  // [768]
    const float* proj_w = (const float*)d_in[4];  // [256,256]
    const float* proj_b = (const float*)d_in[5];  // [256]
    float* y = (float*)d_out;

    // workspace layout (floats); total ~31.5 MB
    float* ws    = (float*)d_ws;
    float* Gpart = ws;                 // 8*8*256*256    = 4194304
    float* G     = Gpart + 4194304;    // 8*256*256      = 524288
    float* Sx    = G + 524288;         // 8*256          = 2048
    float* WqkG  = Sx + 2048;          // 8*512*256      = 1048576
    float* norms = WqkG + 1048576;     // 8*512          = 4096
    float* vvec  = norms + 4096;       // 8*512          = 4096
    float* QKraw = vvec + 4096;        // 8*256*256      = 524288
    float* attn  = QKraw + 524288;     // 8*256*256      = 524288
    float* M1    = attn + 524288;      // 8*256*256      = 524288
    float* Meff  = M1 + 524288;        // 8*256*256      = 524288
    float* beff  = Meff + 524288;      // 8*256          = 2048

    // 1. row sums of X (for bias terms)
    rowsum_kernel<<<dim3(NB * CH), 256, 0, stream>>>(x, Sx);

    // 2. split-K partial Grams (pass 1 over x, WMMA) + deterministic reduce
    gram_kernel<<<dim3(4, 4, NB * KSLICES), 128, 0, stream>>>(x, Gpart);
    greduce_kernel<<<dim3(NB * CH * CH / 256), 256, 0, stream>>>(Gpart, G);

    // 3. WqkG[b] = [Wq;Wk] @ G[b]
    gemm_nn_kernel<<<dim3(CH / 16, 512 / 16, NB), dim3(16, 16), 0, stream>>>(
        qkv_w, G, WqkG, CH, CH, CH, CH, 0L, (long)(CH * CH), (long)(512 * CH));

    // 4. norms + v = W.Sx
    norms_kernel<<<dim3(512, NB), 256, 0, stream>>>(WqkG, qkv_w, qkv_b, Sx, norms, vvec);

    // 5. QKraw[b] = WqG[b] @ Wk^T
    gemm_nt_kernel<<<dim3(CH / 16, CH / 16, NB), dim3(16, 16), 0, stream>>>(
        WqkG, qkv_w + 256 * CH, QKraw, CH, CH, CH, CH, (long)(512 * CH), 0L, (long)(CH * CH));

    // 6. normalize + scale + softmax -> attn
    softmax_kernel<<<dim3(CH, NB), 256, 0, stream>>>(QKraw, norms, vvec, qkv_b, attn);

    // 7. M1[b] = proj_w @ attn[b]
    gemm_nn_kernel<<<dim3(CH / 16, CH / 16, NB), dim3(16, 16), 0, stream>>>(
        proj_w, attn, M1, CH, CH, CH, CH, 0L, (long)(CH * CH), (long)(CH * CH));

    // 8. Meff[b] = M1[b] @ Wv
    gemm_nn_kernel<<<dim3(CH / 16, CH / 16, NB), dim3(16, 16), 0, stream>>>(
        M1, qkv_w + 512 * CH, Meff, CH, CH, CH, CH, (long)(CH * CH), 0L, (long)(CH * CH));

    // 9. beff[b] = M1[b] @ bv + proj_b
    beff_kernel<<<dim3(NB), 256, 0, stream>>>(M1, qkv_b, proj_b, beff);

    // 10. y[b] = Meff[b] @ X[b] + beff[b]  (pass 2 over x, WMMA)
    out_kernel<<<dim3(HW / 128, CH / 64, NB), 256, 0, stream>>>(Meff, x, beff, y);
}
